// GraphIpaFrameDenoisingLayer_31112743092520
// MI455X (gfx1250) — matrix operand order repaired and checked by reading, not compiled, not run on hardware
//
#include <hip/hip_runtime.h>
#include <stdint.h>
#include <math.h>

// ---------------------------------------------------------------------------
// Problem constants (from reference)
// ---------------------------------------------------------------------------
constexpr int kN    = 16384;
constexpr int kE    = 262144;
constexpr int kEseq = 32768;
constexpr int kCS   = 256;
constexpr int kCZ   = 64;
constexpr int kH    = 8;
constexpr int kCH   = 16;
constexpr int kPQK  = 8;
constexpr int kPV   = 12;
constexpr int kHID  = 320;   // 2*(C_S//2) + C_Z

// WMMA vector types (gfx1250, wave32)
typedef __attribute__((ext_vector_type(16))) __bf16 v16bf;
typedef __attribute__((ext_vector_type(8)))  float  v8f;

// ---------------------------------------------------------------------------
// bf16 helpers (bit-level, no __bf16 arithmetic needed)
// ---------------------------------------------------------------------------
union U32BF { unsigned u; __bf16 h[2]; };

__device__ __forceinline__ __bf16 f2bf(float f) {
  union { float f; unsigned u; } x; x.f = f;
  unsigned r = x.u + 0x7FFFu + ((x.u >> 16) & 1u);   // round-to-nearest-even
  union { unsigned short s; __bf16 b; } y;
  y.s = (unsigned short)(r >> 16);
  return y.b;
}
__device__ __forceinline__ float bf2f(__bf16 b) {
  union { unsigned short s; __bf16 b; } x; x.b = b;
  union { unsigned u; float f; } y; y.u = ((unsigned)x.s) << 16;
  return y.f;
}

__device__ __forceinline__ void atomicMaxF(float* addr, float val) {
  int* ia = (int*)addr;
  int old = __float_as_int(*addr);
  while (__int_as_float(old) < val) {
    int assumed = old;
    old = atomicCAS(ia, assumed, __float_as_int(val));
    if (old == assumed) break;
  }
}

// load one wave's A-operand (16x32 bf16 tile row-slice) from a row pointer
__device__ __forceinline__ v16bf load_a_frag(const __bf16* arow, int k0, int half) {
  v16bf a;
  const int a0 = k0 + half * 16;
  const int a1 = k0 + half * 16 + 8;
#pragma unroll
  for (int v = 0; v < 4; ++v) {
    U32BF p; p.u = *(const unsigned*)(arow + a0 + 2 * v);
    a[2 * v] = p.h[0]; a[2 * v + 1] = p.h[1];
  }
#pragma unroll
  for (int v = 0; v < 4; ++v) {
    U32BF p; p.u = *(const unsigned*)(arow + a1 + 2 * v);
    a[8 + 2 * v] = p.h[0]; a[9 + 2 * v] = p.h[1];
  }
  return a;
}
// load one wave's B-operand (32x16) from a transposed-weight column pointer
__device__ __forceinline__ v16bf load_b_frag(const __bf16* bcol, int k0, int half) {
  v16bf b;
  const int bb = k0 + half * 16;
#pragma unroll
  for (int v = 0; v < 8; ++v) {
    U32BF p; p.u = *(const unsigned*)(bcol + bb + 2 * v);
    b[2 * v] = p.h[0]; b[2 * v + 1] = p.h[1];
  }
  return b;
}

// ---------------------------------------------------------------------------
// Register-blocked bf16 WMMA GEMM: C[M,Nd] = A[M,K] * W + bias
// W pre-transposed as BT[Nd,K] bf16. One wave computes a 16 x (16*NT) strip:
// each A fragment is reused for NT WMMAs (cuts A HBM re-fetch by NT).
// M % 64 == 0, Nd % (16*NT) == 0, K % 32 == 0.
// ---------------------------------------------------------------------------
template <int NT>
__global__ void k_gemm(const __bf16* __restrict__ A, const __bf16* __restrict__ BT,
                       const float* __restrict__ bias, float* Cf, __bf16* Cb,
                       int M, int Nd, int K, int relu) {
  const int lane = threadIdx.x;                 // 0..31
  const int tm   = blockIdx.y * blockDim.y + threadIdx.y;
  const int row0 = tm * 16;
  const int col0 = blockIdx.x * 16 * NT;
  if (row0 >= M) return;                        // wave-uniform

  const int r    = row0 + (lane & 15);
  const int cl   = lane & 15;
  const int half = lane >> 4;

  v8f acc[NT];
#pragma unroll
  for (int t = 0; t < NT; ++t) acc[t] = v8f{0.f, 0.f, 0.f, 0.f, 0.f, 0.f, 0.f, 0.f};

  const __bf16* arow = A + (size_t)r * K;
  for (int k0 = 0; k0 < K; k0 += 32) {
    v16bf a = load_a_frag(arow, k0, half);
#pragma unroll
    for (int t = 0; t < NT; ++t) {
      const __bf16* bcol = BT + (size_t)(col0 + t * 16 + cl) * K;
      v16bf b = load_b_frag(bcol, k0, half);
      acc[t] = __builtin_amdgcn_wmma_f32_16x16x32_bf16(false, a, false, b,
                                                       (short)0, acc[t], false, false);
    }
  }
#pragma unroll
  for (int t = 0; t < NT; ++t) {
    const int c = col0 + t * 16 + cl;
    const float bs = bias ? bias[c] : 0.f;
#pragma unroll
    for (int v = 0; v < 8; ++v) {
      float x = acc[t][v] + bs;
      if (relu) x = fmaxf(x, 0.f);
      const int rr = row0 + half * 8 + v;
      if (Cf) Cf[(size_t)rr * Nd + c] = x;
      if (Cb) Cb[(size_t)rr * Nd + c] = f2bf(x);
    }
  }
}

// ---------------------------------------------------------------------------
// Fused edge-transition MLP. One 4-wave block per 16 edges.
//   x = [h[i], h[j], z[e]]           (built in LDS, never hits HBM)
//   t = relu(x @ W0 + b0); t = relu(t @ W1 + b1)
//   y = (t + x) @ Wf + bf; out = LayerNorm(y) * g + b
// Trunk: each wave owns 5 column tiles (4*5*16 = 320). All weights stay in
// L2 (~400 KB); HBM traffic is just z-in + 64-float rows out.
// LDS row strides padded to dodge bank conflicts.
// ---------------------------------------------------------------------------
constexpr int kLSTR = 328;   // 320 + 8 bf16 pad
constexpr int kYSTR = 65;    // 64 + 1 f32 pad

__global__ void k_edge_mlp(const __bf16* __restrict__ hbf, const float* __restrict__ z,
                           const int* __restrict__ ii, const int* __restrict__ jj,
                           const __bf16* __restrict__ W0, const float* __restrict__ b0,
                           const __bf16* __restrict__ W1, const float* __restrict__ b1,
                           const __bf16* __restrict__ Wf, const float* __restrict__ bf_,
                           const float* __restrict__ lng, const float* __restrict__ lnb,
                           float* __restrict__ out, int E) {
  __shared__ __bf16 X [16 * kLSTR];
  __shared__ __bf16 Ta[16 * kLSTR];
  __shared__ __bf16 Tb[16 * kLSTR];
  __shared__ float  Y [16 * kYSTR];

  const int e0   = blockIdx.x * 16;
  const int lane = threadIdx.x;
  const int wy   = threadIdx.y;            // wave id 0..3
  const int tid  = wy * 32 + lane;         // 0..127
  const int cl   = lane & 15;
  const int half = lane >> 4;

  // ---- build X in LDS (gathered, contiguous 256B/256B/128B row chunks) ----
  for (int t = tid; t < 16 * kHID; t += 128) {
    int r = t / kHID, c = t - r * kHID;
    int e = e0 + r;
    __bf16 v;
    if (c < 128)      v = hbf[(size_t)ii[e] * 128 + c];
    else if (c < 256) v = hbf[(size_t)jj[e] * 128 + (c - 128)];
    else              v = f2bf(z[(size_t)e * kCZ + (c - 256)]);
    X[r * kLSTR + c] = v;
  }
  __syncthreads();

  // ---- one trunk layer: LDS src -> LDS dst, 5 column tiles per wave ----
  auto layer = [&](const __bf16* src, const __bf16* WT, const float* bias,
                   __bf16* dst, bool relu) {
    v8f acc[5];
#pragma unroll
    for (int t = 0; t < 5; ++t) acc[t] = v8f{0.f, 0.f, 0.f, 0.f, 0.f, 0.f, 0.f, 0.f};
    const __bf16* arow = src + (size_t)cl * kLSTR;      // lane's row in LDS
    for (int k0 = 0; k0 < kHID; k0 += 32) {
      v16bf a = load_a_frag(arow, k0, half);
#pragma unroll
      for (int t = 0; t < 5; ++t) {
        const int c = wy * 80 + t * 16 + cl;
        v16bf b = load_b_frag(WT + (size_t)c * kHID, k0, half);
        acc[t] = __builtin_amdgcn_wmma_f32_16x16x32_bf16(false, a, false, b,
                                                         (short)0, acc[t], false, false);
      }
    }
#pragma unroll
    for (int t = 0; t < 5; ++t) {
      const int c = wy * 80 + t * 16 + cl;
      const float bs = bias[c];
#pragma unroll
      for (int v = 0; v < 8; ++v) {
        float x = acc[t][v] + bs;
        if (relu) x = fmaxf(x, 0.f);
        dst[(half * 8 + v) * kLSTR + c] = f2bf(x);
      }
    }
  };

  layer(X, W0, b0, Ta, true);
  __syncthreads();
  layer(Ta, W1, b1, Tb, true);
  __syncthreads();

  // ---- residual: Ta = Tb + X ----
  for (int t = tid; t < 16 * kHID; t += 128) {
    int r = t / kHID, c = t - r * kHID;
    Ta[r * kLSTR + c] = f2bf(bf2f(Tb[r * kLSTR + c]) + bf2f(X[r * kLSTR + c]));
  }
  __syncthreads();

  // ---- final 320 -> 64: one 16x16 tile per wave ----
  {
    v8f acc = {0.f, 0.f, 0.f, 0.f, 0.f, 0.f, 0.f, 0.f};
    const __bf16* arow = Ta + (size_t)cl * kLSTR;
    const int c = wy * 16 + cl;
    const __bf16* bcol = Wf + (size_t)c * kHID;
    for (int k0 = 0; k0 < kHID; k0 += 32) {
      v16bf a = load_a_frag(arow, k0, half);
      v16bf b = load_b_frag(bcol, k0, half);
      acc = __builtin_amdgcn_wmma_f32_16x16x32_bf16(false, a, false, b,
                                                    (short)0, acc, false, false);
    }
    const float bs = bf_[c];
#pragma unroll
    for (int v = 0; v < 8; ++v)
      Y[(half * 8 + v) * kYSTR + c] = acc[v] + bs;
  }
  __syncthreads();

  // ---- LayerNorm(64) per row, one thread per row ----
  if (tid < 16) {
    const float* yr = Y + tid * kYSTR;
    float mean = 0.f;
#pragma unroll 8
    for (int c = 0; c < 64; ++c) mean += yr[c];
    mean *= (1.f / 64.f);
    float var = 0.f;
#pragma unroll 8
    for (int c = 0; c < 64; ++c) { float d = yr[c] - mean; var += d * d; }
    float inv = rsqrtf(var * (1.f / 64.f) + 1e-5f);
    float* orow = out + (size_t)(e0 + tid) * 64;
#pragma unroll 8
    for (int c = 0; c < 64; ++c)
      orow[c] = (yr[c] - mean) * inv * lng[c] + lnb[c];
  }
}

// ---------------------------------------------------------------------------
// Elementwise / conversion kernels
// ---------------------------------------------------------------------------
__global__ void k_cvt(__bf16* dst, const float* src, int n) {
  int i = blockIdx.x * blockDim.x + threadIdx.x;
  if (i < n) dst[i] = f2bf(src[i]);
}
// transpose-convert weight: W[K,Nd] f32 -> BT[Nd,K] bf16
__global__ void k_wT(__bf16* dst, const float* W, int K, int Nd, int total) {
  int i = blockIdx.x * blockDim.x + threadIdx.x;
  if (i >= total) return;
  int k = i / Nd, n = i - k * Nd;
  dst[(size_t)n * K + k] = f2bf(W[i]);
}
__global__ void k_fill(float* p, float v, int n) {
  int i = blockIdx.x * blockDim.x + threadIdx.x;
  if (i < n) p[i] = v;
}
__global__ void k_mask(float* s, const float* m, int n) {
  int i = blockIdx.x * blockDim.x + threadIdx.x;
  if (i < n * kCS) s[i] *= m[i / kCS];
}

// ---------------------------------------------------------------------------
// IPA: local->global point transform. qp[N,H,8,3], kvp[N,H,20,3]
// ---------------------------------------------------------------------------
__global__ void k_points(const float* qp, const float* kvp, const float* rots,
                         const float* trans, float* qpts, float* kpts, float* vpts,
                         int n) {
  int idx = blockIdx.x * blockDim.x + threadIdx.x;
  int total = n * kH * 28;
  if (idx >= total) return;
  int nh = idx / 28, p = idx - nh * 28;
  int node = nh / kH, h = nh - node * kH;
  const float* R = rots + (size_t)node * 9;
  const float* T = trans + (size_t)node * 3;
  float l0, l1, l2;
  float* dst;
  if (p < 8) {
    const float* s = qp + (size_t)node * 192 + h * 24 + p * 3;
    l0 = s[0]; l1 = s[1]; l2 = s[2];
    dst = qpts + (size_t)node * 192 + h * 24 + p * 3;
  } else {
    int p2 = p - 8;
    const float* s = kvp + (size_t)node * 480 + h * 60 + p2 * 3;
    l0 = s[0]; l1 = s[1]; l2 = s[2];
    if (p2 < 8) dst = kpts + (size_t)node * 192 + h * 24 + p2 * 3;
    else        dst = vpts + (size_t)node * 288 + h * 36 + (p2 - 8) * 3;
  }
  dst[0] = R[0] * l0 + R[1] * l1 + R[2] * l2 + T[0];
  dst[1] = R[3] * l0 + R[4] * l1 + R[5] * l2 + T[1];
  dst[2] = R[6] * l0 + R[7] * l1 + R[8] * l2 + T[2];
}

// bz[e,h] = z[e,:] @ Wb[:,h] + bb[h]   (K=64, tiny -> VALU)
__global__ void k_edge_bias(const float* z, const float* W, const float* b,
                            float* bz, int E) {
  int idx = blockIdx.x * blockDim.x + threadIdx.x;
  if (idx >= E * kH) return;
  int e = idx / kH, h = idx - e * kH;
  const float* zr = z + (size_t)e * kCZ;
  float acc = b[h];
#pragma unroll 8
  for (int c = 0; c < kCZ; ++c) acc += zr[c] * W[c * kH + h];
  bz[idx] = acc;
}

// Pass 1: logits + segment max
__global__ void k_edge_logits(const float* q, const float* kv, const float* qpts,
                              const float* kpts, const float* bz, const float* hw,
                              const float* mask, const int* ii, const int* jj,
                              float* logits, float* mx, int E) {
  int idx = blockIdx.x * blockDim.x + threadIdx.x;
  if (idx >= E * kH) return;
  int e = idx / kH, h = idx - e * kH;
  int i = ii[e], j = jj[e];
  const float* qr = q   + (size_t)i * 128 + h * 16;
  const float* kr = kv  + (size_t)j * 256 + h * 32;
  float qk = 0.f;
#pragma unroll
  for (int c = 0; c < 16; ++c) qk += qr[c] * kr[c];
  qk *= 0.14433756729740643f;                       // sqrt(1/(3*C_H))
  const float* qp = qpts + (size_t)i * 192 + h * 24;
  const float* kp = kpts + (size_t)j * 192 + h * 24;
  float d2 = 0.f;
#pragma unroll
  for (int t = 0; t < 24; ++t) { float d = qp[t] - kp[t]; d2 += d * d; }
  float x = hw[h];
  float sp = (x > 20.f) ? x : log1pf(expf(x));      // softplus(head_weights)
  float pt = -0.5f * 0.09622504486493764f * sp * d2; // 0.5*sqrt(1/108)
  float lg = qk + 0.5773502691896258f * bz[idx] + pt + 1e9f * (mask[j] - 1.f);
  logits[idx] = lg;
  atomicMaxF(&mx[(size_t)i * kH + h], lg);
}

// Pass 2: exp + segment sum (overwrites logits with ex)
__global__ void k_edge_exp(float* logits, const float* mx, float* den,
                           const int* ii, int E) {
  int idx = blockIdx.x * blockDim.x + threadIdx.x;
  if (idx >= E * kH) return;
  int e = idx / kH, h = idx - e * kH;
  int i = ii[e];
  float m = fmaxf(mx[(size_t)i * kH + h], -1e30f);
  float ex = expf(logits[idx] - m);
  logits[idx] = ex;
  atomicAdd(&den[(size_t)i * kH + h], ex);
}

// Pass 3: weighted scatter-accumulate (o, o_pt_g, o_pair)
__global__ void k_edge_accum(const float* ex, const float* den, const float* kv,
                             const float* vpts, const float* z, const int* ii,
                             const int* jj, float* o, float* optg, float* opair,
                             int E) {
  int idx = blockIdx.x * blockDim.x + threadIdx.x;
  if (idx >= E * kH) return;
  int e = idx / kH, h = idx - e * kH;
  int i = ii[e], j = jj[e];
  float a = ex[idx] / (den[(size_t)i * kH + h] + 1e-8f);
  const float* vr = kv + (size_t)j * 256 + h * 32 + 16;
  float* od = o + (size_t)i * 128 + h * 16;
#pragma unroll
  for (int c = 0; c < 16; ++c) atomicAdd(&od[c], a * vr[c]);
  const float* vp = vpts + (size_t)j * 288 + h * 36;
  float* og = optg + (size_t)i * 288 + h * 36;
#pragma unroll
  for (int t = 0; t < 36; ++t) atomicAdd(&og[t], a * vp[t]);
  const float* zr = z + (size_t)e * kCZ;
  float* op = opair + (size_t)i * 512 + h * 64;
#pragma unroll 8
  for (int c = 0; c < 64; ++c) atomicAdd(&op[c], a * zr[c]);
}

// Finalize: o_pt = R^T (o_pt_g - t), norms, concat feats -> bf16 [N,1024]
__global__ void k_finalize(const float* o, const float* optg, const float* opair,
                           const float* rots, const float* trans, __bf16* feats,
                           int n) {
  int idx = blockIdx.x * blockDim.x + threadIdx.x;
  if (idx >= n * kH) return;
  int node = idx / kH, h = idx - node * kH;
  const float* R = rots + (size_t)node * 9;
  const float* T = trans + (size_t)node * 3;
  __bf16* f = feats + (size_t)node * 1024;
  const float* orow = o + (size_t)node * 128 + h * 16;
#pragma unroll
  for (int c = 0; c < 16; ++c) f[h * 16 + c] = f2bf(orow[c]);
#pragma unroll
  for (int p = 0; p < kPV; ++p) {
    const float* g = optg + (size_t)node * 288 + h * 36 + p * 3;
    float g0 = g[0] - T[0], g1 = g[1] - T[1], g2 = g[2] - T[2];
    float l0 = R[0] * g0 + R[3] * g1 + R[6] * g2;   // R^T g
    float l1 = R[1] * g0 + R[4] * g1 + R[7] * g2;
    float l2 = R[2] * g0 + R[5] * g1 + R[8] * g2;
    f[128 + h * 36 + p * 3 + 0] = f2bf(l0);
    f[128 + h * 36 + p * 3 + 1] = f2bf(l1);
    f[128 + h * 36 + p * 3 + 2] = f2bf(l2);
    f[416 + h * 12 + p] = f2bf(sqrtf(l0 * l0 + l1 * l1 + l2 * l2 + 1e-8f));
  }
  const float* pr = opair + (size_t)node * 512 + h * 64;
#pragma unroll 8
  for (int c = 0; c < 64; ++c) f[512 + h * 64 + c] = f2bf(pr[c]);
}

// out = LayerNorm(s + u*m) over 256 dims, one wave32 per row
__global__ void k_residual_ln(const float* s, const float* u, const float* mask,
                              const float* g, const float* b, float* out, int n) {
  int row = blockIdx.x * blockDim.y + threadIdx.y;
  if (row >= n) return;
  int lane = threadIdx.x;
  float m = mask ? mask[row] : 1.f;
  float vals[8];
  float sum = 0.f;
#pragma unroll
  for (int t = 0; t < 8; ++t) {
    int c = lane + t * 32;
    float x = s[(size_t)row * kCS + c];
    if (u) x += u[(size_t)row * kCS + c] * m;
    vals[t] = x; sum += x;
  }
  for (int o = 16; o; o >>= 1) sum += __shfl_xor(sum, o, 32);
  float mean = sum * (1.f / kCS);
  float var = 0.f;
#pragma unroll
  for (int t = 0; t < 8; ++t) { float d = vals[t] - mean; var += d * d; }
  for (int o = 16; o; o >>= 1) var += __shfl_xor(var, o, 32);
  float inv = rsqrtf(var * (1.f / kCS) + 1e-5f);
#pragma unroll
  for (int t = 0; t < 8; ++t) {
    int c = lane + t * 32;
    out[(size_t)row * kCS + c] = (vals[t] - mean) * inv * g[c] + b[c];
  }
}

// backbone update + quaternion compose (per node)
__global__ void k_bb_compose(const float* s, const float* nm, const float* W,
                             const float* B, const float* rots, const float* trans,
                             float* out_r, float* out_t, int n) {
  int node = blockIdx.x * blockDim.x + threadIdx.x;
  if (node >= n) return;
  const float* sr = s + (size_t)node * kCS;
  float m = nm[node];
  float up[6];
#pragma unroll
  for (int o6 = 0; o6 < 6; ++o6) {
    float acc = B[o6];
    for (int c = 0; c < kCS; ++c) acc += sr[c] * m * W[c * 6 + o6];
    up[o6] = acc * m;
  }
  float w = 1.f, x = up[0], y = up[1], z = up[2];
  float qn = rsqrtf(w * w + x * x + y * y + z * z);
  w *= qn; x *= qn; y *= qn; z *= qn;
  float ru[9];
  ru[0] = w * w + x * x - y * y - z * z; ru[1] = 2 * (x * y - w * z); ru[2] = 2 * (x * z + w * y);
  ru[3] = 2 * (x * y + w * z); ru[4] = w * w - x * x + y * y - z * z; ru[5] = 2 * (y * z - w * x);
  ru[6] = 2 * (x * z - w * y); ru[7] = 2 * (y * z + w * x); ru[8] = w * w - x * x - y * y + z * z;
  const float* R = rots + (size_t)node * 9;
  float* Ro = out_r + (size_t)node * 9;
#pragma unroll
  for (int a = 0; a < 3; ++a)
#pragma unroll
    for (int c = 0; c < 3; ++c)
      Ro[a * 3 + c] = R[a * 3 + 0] * ru[0 * 3 + c] + R[a * 3 + 1] * ru[1 * 3 + c] + R[a * 3 + 2] * ru[2 * 3 + c];
  const float* T = trans + (size_t)node * 3;
  float* To = out_t + (size_t)node * 3;
#pragma unroll
  for (int a = 0; a < 3; ++a)
    To[a] = R[a * 3 + 0] * up[3] + R[a * 3 + 1] * up[4] + R[a * 3 + 2] * up[5] + T[a];
}

// ---------------------------------------------------------------------------
// Host orchestration
// ---------------------------------------------------------------------------
// Assumed d_in layout: the 9 arrays in setup_inputs() dict order, then the
// params pytree leaves in jax.tree_util order (dict keys sorted, lists in
// order). See index map below.
enum {
  IN_NODE = 0, IN_ROTS, IN_TRANS, IN_EDGEF, IN_EIDX, IN_SEQEF, IN_SEIDX,
  IN_MASK, IN_NMASK,
  P_BB_B = 9, P_BB_W = 10,
  P_ET = 11,       // +0 fin.b +1 fin.w +2 init.b +3 init.w +4 ln.b +5 ln.g
                   // +6 t0.b +7 t0.w +8 t1.b +9 t1.w
  P_IPA_SEQ = 21,  // +0 b.b +1 b.w +2 hw +3 kv.b +4 kv.w +5 kvp.b +6 kvp.w
                   // +7 out.b +8 out.w +9 q.b +10 q.w +11 qp.b +12 qp.w
  P_IPA_SP = 34,
  P_LN1 = 47, P_LN2 = 49,        // b, g
  P_NT = 51,       // l1.b l1.w l2.b l2.w l3.b l3.w ln.b ln.g
  P_SET = 59
};

struct IpaW { __bf16 *qw, *kvw, *qpw, *kvpw, *outw; };
struct EtW  { __bf16 *initw, *t0w, *t1w, *finw; };

extern "C" void kernel_launch(void* const* d_in, const int* in_sizes, int n_in,
                              void* d_out, int out_size, void* d_ws, size_t ws_size,
                              hipStream_t stream) {
  (void)in_sizes; (void)n_in; (void)out_size; (void)ws_size;
  const float* nodef = (const float*)d_in[IN_NODE];
  const float* rots  = (const float*)d_in[IN_ROTS];
  const float* trans = (const float*)d_in[IN_TRANS];
  const float* edgef = (const float*)d_in[IN_EDGEF];
  const int*   eidx  = (const int*)d_in[IN_EIDX];
  const float* seqef = (const float*)d_in[IN_SEQEF];
  const int*   seidx = (const int*)d_in[IN_SEIDX];
  const float* mask  = (const float*)d_in[IN_MASK];
  const float* nmask = (const float*)d_in[IN_NMASK];
  auto P = [&](int i) { return (const float*)d_in[i]; };

  char* ws = (char*)d_ws;
  size_t cur = 0;
  auto alloc = [&](size_t bytes) {
    cur = (cur + 255) & ~(size_t)255;
    size_t o = cur; cur += bytes; return o;
  };
  auto ew = [&](int blocks) { return dim3(blocks); };
  const dim3 T256(256);

  // ---- weight conversion (f32 [K,Nd] -> bf16 transposed [Nd,K]) ----
  auto convW = [&](int pidx, int K, int Nd) -> __bf16* {
    size_t off = alloc((size_t)K * Nd * 2);
    __bf16* dst = (__bf16*)(ws + off);
    int total = K * Nd;
    k_wT<<<ew((total + 255) / 256), T256, 0, stream>>>(dst, P(pidx), K, Nd, total);
    return dst;
  };
  auto mkIpa = [&](int base) {
    IpaW w;
    w.qw   = convW(base + 10, kCS, 128);
    w.kvw  = convW(base + 4,  kCS, 256);
    w.qpw  = convW(base + 12, kCS, 192);
    w.kvpw = convW(base + 6,  kCS, 480);
    w.outw = convW(base + 8,  1024, kCS);
    return w;
  };
  auto mkEt = [&](int base) {
    EtW w;
    w.initw = convW(base + 3, kCS, 128);
    w.t0w   = convW(base + 7, kHID, kHID);
    w.t1w   = convW(base + 9, kHID, kHID);
    w.finw  = convW(base + 1, kHID, 64);
    return w;
  };
  IpaW Wsp  = mkIpa(P_IPA_SP);
  IpaW Wseq = mkIpa(P_IPA_SEQ);
  __bf16* l1w = convW(P_NT + 1, kCS, kCS);
  __bf16* l2w = convW(P_NT + 3, kCS, kCS);
  __bf16* l3w = convW(P_NT + 5, kCS, kCS);
  EtW Wet  = mkEt(P_ET);
  EtW Wset = mkEt(P_SET);

  // ---- persistent activation buffers ----
  float*  s1      = (float*) (ws + alloc((size_t)kN * kCS * 4));
  float*  s2      = (float*) (ws + alloc((size_t)kN * kCS * 4));
  float*  ubuf    = (float*) (ws + alloc((size_t)kN * kCS * 4));
  __bf16* sbf     = (__bf16*)(ws + alloc((size_t)kN * kCS * 2));
  __bf16* featsbf = (__bf16*)(ws + alloc((size_t)kN * 1024 * 2));
  __bf16* nbf1    = (__bf16*)(ws + alloc((size_t)kN * kCS * 2));
  __bf16* nbf2    = (__bf16*)(ws + alloc((size_t)kN * kCS * 2));
  __bf16* hbf     = (__bf16*)(ws + alloc((size_t)kN * 128 * 2));

  // ---- scratch region (IPA passes; phases strictly sequential) ----
  float* qb    = (float*)(ws + alloc((size_t)kN * 128 * 4));
  float* kvb   = (float*)(ws + alloc((size_t)kN * 256 * 4));
  float* qpb   = (float*)(ws + alloc((size_t)kN * 192 * 4));
  float* kvpb  = (float*)(ws + alloc((size_t)kN * 480 * 4));
  float* qpts  = (float*)(ws + alloc((size_t)kN * 192 * 4));
  float* kpts  = (float*)(ws + alloc((size_t)kN * 192 * 4));
  float* vpts  = (float*)(ws + alloc((size_t)kN * 288 * 4));
  float* ob    = (float*)(ws + alloc((size_t)kN * 128 * 4));
  float* optg  = (float*)(ws + alloc((size_t)kN * 288 * 4));
  float* opair = (float*)(ws + alloc((size_t)kN * 512 * 4));
  float* mxb   = (float*)(ws + alloc((size_t)kN * kH * 4));
  float* denb  = (float*)(ws + alloc((size_t)kN * kH * 4));
  float* logb  = (float*)(ws + alloc((size_t)kE * kH * 4));
  float* bzb   = (float*)(ws + alloc((size_t)kE * kH * 4));

  // ---- launch helpers ----
  auto gemm = [&](const __bf16* A, const __bf16* BT, const float* bias,
                  float* Cf, __bf16* Cb, int M, int Nd, int K, int relu) {
    if (Nd % 64 == 0) {
      dim3 grid(Nd / 64, M / 64), blk(32, 4);
      k_gemm<4><<<grid, blk, 0, stream>>>(A, BT, bias, Cf, Cb, M, Nd, K, relu);
    } else {
      dim3 grid(Nd / 32, M / 64), blk(32, 4);
      k_gemm<2><<<grid, blk, 0, stream>>>(A, BT, bias, Cf, Cb, M, Nd, K, relu);
    }
  };
  auto cvt = [&](__bf16* dst, const float* src, int n) {
    k_cvt<<<ew((n + 255) / 256), T256, 0, stream>>>(dst, src, n);
  };
  auto fill = [&](float* p, float v, int n) {
    k_fill<<<ew((n + 255) / 256), T256, 0, stream>>>(p, v, n);
  };

  // ---- one IPA block ----
  auto run_ipa = [&](const float* sin, int base, IpaW& W, const int* ei,
                     int Ecnt, const float* zf, float* uout) {
    const int* ii = ei;
    const int* jj = ei + Ecnt;
    cvt(sbf, sin, kN * kCS);
    gemm(sbf, W.qw,   P(base + 9),  qb,   nullptr, kN, 128, kCS, 0);
    gemm(sbf, W.kvw,  P(base + 3),  kvb,  nullptr, kN, 256, kCS, 0);
    gemm(sbf, W.qpw,  P(base + 11), qpb,  nullptr, kN, 192, kCS, 0);
    gemm(sbf, W.kvpw, P(base + 5),  kvpb, nullptr, kN, 480, kCS, 0);
    {
      int tot = kN * kH * 28;
      k_points<<<ew((tot + 255) / 256), T256, 0, stream>>>(
          qpb, kvpb, rots, trans, qpts, kpts, vpts, kN);
    }
    k_edge_bias<<<ew((Ecnt * kH + 255) / 256), T256, 0, stream>>>(
        zf, P(base + 1), P(base + 0), bzb, Ecnt);
    fill(mxb, -1e30f, kN * kH);
    fill(denb, 0.f, kN * kH);
    fill(ob, 0.f, kN * 128);
    fill(optg, 0.f, kN * 288);
    fill(opair, 0.f, kN * 512);
    int eh = Ecnt * kH;
    k_edge_logits<<<ew((eh + 255) / 256), T256, 0, stream>>>(
        qb, kvb, qpts, kpts, bzb, P(base + 2), mask, ii, jj, logb, mxb, Ecnt);
    k_edge_exp<<<ew((eh + 255) / 256), T256, 0, stream>>>(logb, mxb, denb, ii, Ecnt);
    k_edge_accum<<<ew((eh + 255) / 256), T256, 0, stream>>>(
        logb, denb, kvb, vpts, zf, ii, jj, ob, optg, opair, Ecnt);
    k_finalize<<<ew((kN * kH + 255) / 256), T256, 0, stream>>>(
        ob, optg, opair, rots, trans, featsbf, kN);
    gemm(featsbf, W.outw, P(base + 7), uout, nullptr, kN, 256, 1024, 0);
  };

  // ========================= pipeline =========================
  // IPA (spatial edges) + residual LN1
  run_ipa(nodef, P_IPA_SP, Wsp, eidx, kE, edgef, ubuf);
  k_residual_ln<<<dim3(kN / 8), dim3(32, 8), 0, stream>>>(
      nodef, ubuf, mask, P(P_LN1 + 1), P(P_LN1 + 0), s1, kN);

  // IPA (sequence edges) + residual LN2
  run_ipa(s1, P_IPA_SEQ, Wseq, seidx, kEseq, seqef, ubuf);
  k_residual_ln<<<dim3(kN / 8), dim3(32, 8), 0, stream>>>(
      s1, ubuf, mask, P(P_LN2 + 1), P(P_LN2 + 0), s2, kN);

  // node transition: LN(s + MLP(s))
  cvt(sbf, s2, kN * kCS);
  gemm(sbf,  l1w, P(P_NT + 0), nullptr, nbf1, kN, kCS, kCS, 1);
  gemm(nbf1, l2w, P(P_NT + 2), nullptr, nbf2, kN, kCS, kCS, 1);
  gemm(nbf2, l3w, P(P_NT + 4), ubuf,    nullptr, kN, kCS, kCS, 0);
  k_residual_ln<<<dim3(kN / 8), dim3(32, 8), 0, stream>>>(
      s2, ubuf, nullptr, P(P_NT + 7), P(P_NT + 6), s1, kN);

  // s = s * m  -> output segment 0
  k_mask<<<ew((kN * kCS + 255) / 256), T256, 0, stream>>>(s1, mask, kN);
  float* outp = (float*)d_out;
  const size_t O_S = 0;
  const size_t O_R = O_S + (size_t)kN * kCS;
  const size_t O_T = O_R + (size_t)kN * 9;
  const size_t O_E = O_T + (size_t)kN * 3;
  const size_t O_SE = O_E + (size_t)kE * kCZ;
  hipMemcpyAsync(outp + O_S, s1, (size_t)kN * kCS * 4,
                 hipMemcpyDeviceToDevice, stream);

  // backbone update -> new_rots, new_trans
  k_bb_compose<<<ew((kN + 63) / 64), dim3(64), 0, stream>>>(
      s1, nmask, P(P_BB_W), P(P_BB_B), rots, trans, outp + O_R, outp + O_T, kN);

  // edge transitions (spatial + sequence): init projection + fused MLP
  auto run_et = [&](int base, EtW& W, const int* ei, int Ecnt, const float* zf,
                    float* outseg) {
    cvt(sbf, s1, kN * kCS);
    gemm(sbf, W.initw, P(base + 2), nullptr, hbf, kN, 128, kCS, 0);
    k_edge_mlp<<<dim3(Ecnt / 16), dim3(32, 4), 0, stream>>>(
        hbf, zf, ei, ei + Ecnt,
        W.t0w, P(base + 6), W.t1w, P(base + 8),
        W.finw, P(base + 0), P(base + 5), P(base + 4), outseg, Ecnt);
  };
  run_et(P_ET,  Wet,  eidx,  kE,    edgef, outp + O_E);
  run_et(P_SET, Wset, seidx, kEseq, seqef, outp + O_SE);
}
